// DeltaTemporalModel_38457137168810
// MI455X (gfx1250) — matrix operand order repaired and checked by reading, not compiled, not run on hardware
//
#include <hip/hip_runtime.h>
#include <hip/hip_bf16.h>
#include <math.h>

// ---------------- types / constants ----------------
typedef __attribute__((ext_vector_type(16))) __bf16 v16bf;
typedef __attribute__((ext_vector_type(8)))  float  v8f;

static constexpr int kV  = 50257;
static constexpr int kE  = 1024;
static constexpr int kB  = 1024;
static constexpr int kHH = 1024;   // hidden of frozen core MLP
static constexpr int kHD = 4096;   // 4*E
static constexpr int kVMain = 50240;  // 785 * 64 (aligned part of V)

// ---------------- helpers ----------------
__device__ __forceinline__ unsigned short f32_to_bf16_bits(float f) {
  union { float f; unsigned int u; } x; x.f = f;
  unsigned int r = x.u + 0x7FFFu + ((x.u >> 16) & 1u);   // round-to-nearest-even
  return (unsigned short)(r >> 16);
}

__device__ __forceinline__ unsigned int pack_bf16x2(float lo, float hi) {
  return (unsigned int)f32_to_bf16_bits(lo) |
         ((unsigned int)f32_to_bf16_bits(hi) << 16);
}

__device__ __forceinline__ float block_sum(float v, float* red) {
  int tid = threadIdx.x;
  red[tid] = v;
  __syncthreads();
  for (int s = 128; s > 0; s >>= 1) {
    if (tid < s) red[tid] += red[tid + s];
    __syncthreads();
  }
  float r = red[0];
  __syncthreads();
  return r;
}

union FragBf16 {
  v16bf v;
  unsigned short u[16];
  unsigned long long d[4];
};

#define TBM 128
#define TBN 64
#define TBK 32
#define LDSA 36   // shorts per As row (72B, 8B-aligned)
#define LDSB 36   // shorts per Bs row (fp32-source variant)
#define LDSBT 40  // shorts per Bs row for async bf16 variant (80B, 16B-aligned)

// ---------------- tiled bf16 WMMA GEMM (fp32 sources) ----------------
// C[M,N] = A[M,K] @ B[K,N].  TA: A stored [K,M].  TB: B stored [N,K].
// GUARD: per-element bounds checks.  LDC: compile-time C leading dim.
template <bool TA, bool TB, bool GUARD, int LDC>
__global__ __launch_bounds__(256) void gemm_wmma_bf16(
    const float* __restrict__ Ap, const float* __restrict__ Bp,
    float* __restrict__ Cp, int M, int N, int K, int lda, int ldb) {
  __shared__ __attribute__((aligned(16))) unsigned short As[TBM * LDSA]; // [m][k]
  __shared__ __attribute__((aligned(16))) unsigned short Bs[TBN * LDSB]; // [n][k]

  const int tid  = threadIdx.x;
  const int lane = tid & 31;
  const int wave = tid >> 5;
  const int wrow = wave >> 1;
  const int wcol = wave & 1;
  const int hf   = lane >> 4;
  const int l16  = lane & 15;

  const int m0 = blockIdx.y * TBM;
  const int n0 = blockIdx.x * TBN;

  v8f acc[2][2] = {};

  for (int k0 = 0; k0 < K; k0 += TBK) {
    if (GUARD) {
      for (int idx = tid; idx < TBM * TBK; idx += 256) {
        int m = idx >> 5, k = idx & 31;
        int gm = m0 + m, gk = k0 + k;
        float v = 0.0f;
        if (gm < M && gk < K)
          v = TA ? Ap[(size_t)gk * lda + gm] : Ap[(size_t)gm * lda + gk];
        As[m * LDSA + k] = f32_to_bf16_bits(v);
      }
      for (int idx = tid; idx < TBK * TBN; idx += 256) {
        int k = idx >> 6, n = idx & 63;
        int gk = k0 + k, gn = n0 + n;
        float v = 0.0f;
        if (gk < K && gn < N)
          v = TB ? Bp[(size_t)gn * ldb + gk] : Bp[(size_t)gk * ldb + gn];
        Bs[n * LDSB + k] = f32_to_bf16_bits(v);
      }
    } else {
      if (!TA) {
#pragma unroll
        for (int i = 0; i < 4; ++i) {
          int f = tid + i * 256;
          int m = f >> 3;
          int kq = (f & 7) << 2;
          float4 v = *(const float4*)(Ap + (size_t)(m0 + m) * lda + k0 + kq);
          uint2 pk;
          pk.x = pack_bf16x2(v.x, v.y);
          pk.y = pack_bf16x2(v.z, v.w);
          *(uint2*)(&As[m * LDSA + kq]) = pk;
        }
      } else {
#pragma unroll
        for (int i = 0; i < 4; ++i) {
          int f = tid + i * 256;
          int mq = (f & 31) << 2;
          int k = f >> 5;
          float4 v = *(const float4*)(Ap + (size_t)(k0 + k) * lda + m0 + mq);
          As[(mq + 0) * LDSA + k] = f32_to_bf16_bits(v.x);
          As[(mq + 1) * LDSA + k] = f32_to_bf16_bits(v.y);
          As[(mq + 2) * LDSA + k] = f32_to_bf16_bits(v.z);
          As[(mq + 3) * LDSA + k] = f32_to_bf16_bits(v.w);
        }
      }
      if (!TB) {
#pragma unroll
        for (int i = 0; i < 2; ++i) {
          int f = tid + i * 256;
          int nq = (f & 15) << 2;
          int k = f >> 4;
          float4 v = *(const float4*)(Bp + (size_t)(k0 + k) * ldb + n0 + nq);
          Bs[(nq + 0) * LDSB + k] = f32_to_bf16_bits(v.x);
          Bs[(nq + 1) * LDSB + k] = f32_to_bf16_bits(v.y);
          Bs[(nq + 2) * LDSB + k] = f32_to_bf16_bits(v.z);
          Bs[(nq + 3) * LDSB + k] = f32_to_bf16_bits(v.w);
        }
      } else {
#pragma unroll
        for (int i = 0; i < 2; ++i) {
          int f = tid + i * 256;
          int kq = (f & 7) << 2;
          int n = f >> 3;
          float4 v = *(const float4*)(Bp + (size_t)(n0 + n) * ldb + k0 + kq);
          uint2 pk;
          pk.x = pack_bf16x2(v.x, v.y);
          pk.y = pack_bf16x2(v.z, v.w);
          *(uint2*)(&Bs[n * LDSB + kq]) = pk;
        }
      }
    }
    __syncthreads();

    FragBf16 afr[2], bfr[2];
#pragma unroll
    for (int t = 0; t < 2; ++t) {
      const unsigned short* pa = &As[(wrow * 32 + t * 16 + l16) * LDSA];
      afr[t].d[0] = *(const unsigned long long*)(pa + hf * 8);
      afr[t].d[1] = *(const unsigned long long*)(pa + hf * 8 + 4);
      afr[t].d[2] = *(const unsigned long long*)(pa + 16 + hf * 8);
      afr[t].d[3] = *(const unsigned long long*)(pa + 16 + hf * 8 + 4);
      const unsigned short* pb = &Bs[(wcol * 32 + t * 16 + l16) * LDSB];
      bfr[t].d[0] = *(const unsigned long long*)(pb + hf * 8);
      bfr[t].d[1] = *(const unsigned long long*)(pb + hf * 8 + 4);
      bfr[t].d[2] = *(const unsigned long long*)(pb + 16 + hf * 8);
      bfr[t].d[3] = *(const unsigned long long*)(pb + 16 + hf * 8 + 4);
    }

#pragma unroll
    for (int ti = 0; ti < 2; ++ti)
#pragma unroll
      for (int tj = 0; tj < 2; ++tj)
        acc[ti][tj] = __builtin_amdgcn_wmma_f32_16x16x32_bf16(
            false, afr[ti].v, false, bfr[tj].v, (short)0, acc[ti][tj],
            false, false);
    __syncthreads();
  }

  // epilogue: one base pointer, compile-time strides fold into store offsets
  float* cb = Cp + (size_t)(m0 + wrow * 32 + hf * 8) * LDC +
              (n0 + wcol * 32 + l16);
#pragma unroll
  for (int ti = 0; ti < 2; ++ti)
#pragma unroll
    for (int tj = 0; tj < 2; ++tj)
#pragma unroll
      for (int r = 0; r < 8; ++r) {
        if (!GUARD || ((m0 + wrow * 32 + hf * 8 + ti * 16 + r) < M &&
                       (n0 + wcol * 32 + l16 + tj * 16) < N))
          cb[(size_t)(ti * 16 + r) * LDC + tj * 16] = acc[ti][tj][r];
      }
}

// ---------------- LM-head GEMM: A fp32 [M,K], B bf16 [N,K] via async LDS ----
// logits[M, kV] = A @ B^T.  lda = ldb = kE, C leading dim = kV (compile time).
__global__ __launch_bounds__(256) void gemm_wmma_lmhead(
    const float* __restrict__ Ap, const unsigned short* __restrict__ Bw,
    float* __restrict__ Cp) {
  __shared__ __attribute__((aligned(16))) unsigned short As[TBM * LDSA];  // [m][k]
  __shared__ __attribute__((aligned(16))) unsigned short Bs[TBN * LDSBT]; // [n][k]

  const int tid  = threadIdx.x;
  const int lane = tid & 31;
  const int wave = tid >> 5;
  const int wrow = wave >> 1;
  const int wcol = wave & 1;
  const int hf   = lane >> 4;
  const int l16  = lane & 15;

  const int m0 = blockIdx.y * TBM;
  const int n0 = blockIdx.x * TBN;

  // per-thread async-copy slot: 64 rows x 4 chunks of 16B (= 64B/row of bf16)
  const int bn = tid >> 2;              // 0..63
  const int bk = (tid & 3) << 3;        // k element offset: 0,8,16,24
  const unsigned ldsDst =
      (unsigned)(unsigned long long)(uintptr_t)(&Bs[bn * LDSBT + bk]);

  v8f acc[2][2] = {};

  for (int k0 = 0; k0 < kE; k0 += TBK) {
    // A staging: float4 loads + packed bf16 LDS stores
#pragma unroll
    for (int i = 0; i < 4; ++i) {
      int f = tid + i * 256;
      int m = f >> 3;
      int kq = (f & 7) << 2;
      float4 v = *(const float4*)(Ap + (size_t)(m0 + m) * kE + k0 + kq);
      uint2 pk;
      pk.x = pack_bf16x2(v.x, v.y);
      pk.y = pack_bf16x2(v.z, v.w);
      *(uint2*)(&As[m * LDSA + kq]) = pk;
    }
    // B staging: one 16-byte async global->LDS copy per thread (ASYNCcnt)
    {
      const unsigned short* gsrc = Bw + (size_t)(n0 + bn) * kE + k0 + bk;
      unsigned long long gaddr = (unsigned long long)(uintptr_t)gsrc;
      asm volatile("global_load_async_to_lds_b128 %0, %1, off"
                   :: "v"(ldsDst), "v"(gaddr)
                   : "memory");
    }
    asm volatile("s_wait_asynccnt 0x0" ::: "memory");
    __syncthreads();

    FragBf16 afr[2], bfr[2];
#pragma unroll
    for (int t = 0; t < 2; ++t) {
      const unsigned short* pa = &As[(wrow * 32 + t * 16 + l16) * LDSA];
      afr[t].d[0] = *(const unsigned long long*)(pa + hf * 8);
      afr[t].d[1] = *(const unsigned long long*)(pa + hf * 8 + 4);
      afr[t].d[2] = *(const unsigned long long*)(pa + 16 + hf * 8);
      afr[t].d[3] = *(const unsigned long long*)(pa + 16 + hf * 8 + 4);
      const unsigned short* pb =
          &Bs[(wcol * 32 + t * 16 + l16) * LDSBT];
      bfr[t].d[0] = *(const unsigned long long*)(pb + hf * 8);
      bfr[t].d[1] = *(const unsigned long long*)(pb + hf * 8 + 4);
      bfr[t].d[2] = *(const unsigned long long*)(pb + 16 + hf * 8);
      bfr[t].d[3] = *(const unsigned long long*)(pb + 16 + hf * 8 + 4);
    }

#pragma unroll
    for (int ti = 0; ti < 2; ++ti)
#pragma unroll
      for (int tj = 0; tj < 2; ++tj)
        acc[ti][tj] = __builtin_amdgcn_wmma_f32_16x16x32_bf16(
            false, afr[ti].v, false, bfr[tj].v, (short)0, acc[ti][tj],
            false, false);
    __syncthreads();
  }

  float* cb = Cp + (size_t)(m0 + wrow * 32 + hf * 8) * kV +
              (n0 + wcol * 32 + l16);
#pragma unroll
  for (int ti = 0; ti < 2; ++ti)
#pragma unroll
    for (int tj = 0; tj < 2; ++tj)
#pragma unroll
      for (int r = 0; r < 8; ++r)
        cb[(size_t)(ti * 16 + r) * kV + tj * 16] = acc[ti][tj][r];
}

// ---------------- fp32 -> bf16 bulk convert (emb_w), 8 elems/thread --------
__global__ __launch_bounds__(256) void convert_bf16_k(
    const float* __restrict__ in, unsigned short* __restrict__ out,
    long long n8) {  // n8 = n/8
  long long i = (long long)blockIdx.x * 256 + threadIdx.x;
  if (i >= n8) return;
  size_t base = (size_t)i * 8;
  float4 a = *(const float4*)(in + base);
  float4 b = *(const float4*)(in + base + 4);
  uint4 r;
  r.x = pack_bf16x2(a.x, a.y);
  r.y = pack_bf16x2(a.z, a.w);
  r.z = pack_bf16x2(b.x, b.y);
  r.w = pack_bf16x2(b.z, b.w);
  *(uint4*)(out + base) = r;
}

// ---------------- row-wise / elementwise kernels ----------------
__global__ __launch_bounds__(256) void embed_l2norm_k(
    const int* __restrict__ tok, const float* __restrict__ emb_w,
    float* __restrict__ out) {
  __shared__ float red[256];
  const int b = blockIdx.x, tid = threadIdx.x;
  const float* src = emb_w + (size_t)tok[b] * kE;
  float x[4], ss = 0.0f;
#pragma unroll
  for (int i = 0; i < 4; ++i) { x[i] = src[tid + i * 256]; ss += x[i] * x[i]; }
  ss = block_sum(ss, red);
  float inv = 1.0f / fmaxf(sqrtf(ss), 1e-12f);
#pragma unroll
  for (int i = 0; i < 4; ++i) out[(size_t)b * kE + tid + i * 256] = x[i] * inv;
}

__global__ void bias_tanh_k(const float* __restrict__ in,
                            const float* __restrict__ bias,
                            float* __restrict__ out, int n) {
  int i = blockIdx.x * blockDim.x + threadIdx.x;
  if (i < n) out[i] = tanhf(in[i] + bias[i & (kHH - 1)]);
}

__global__ __launch_bounds__(256) void bias_l2norm_k(
    const float* __restrict__ in, const float* __restrict__ bias,
    float* __restrict__ out) {
  __shared__ float red[256];
  const int b = blockIdx.x, tid = threadIdx.x;
  float x[4], ss = 0.0f;
#pragma unroll
  for (int i = 0; i < 4; ++i) {
    int e = tid + i * 256;
    x[i] = in[(size_t)b * kE + e] + bias[e];
    ss += x[i] * x[i];
  }
  ss = block_sum(ss, red);
  float inv = 1.0f / fmaxf(sqrtf(ss), 1e-12f);
#pragma unroll
  for (int i = 0; i < 4; ++i) out[(size_t)b * kE + tid + i * 256] = x[i] * inv;
}

__global__ void sub_k(const float* __restrict__ a, const float* __restrict__ b,
                      float* __restrict__ c, int n) {
  int i = blockIdx.x * blockDim.x + threadIdx.x;
  if (i < n) c[i] = a[i] - b[i];
}

__global__ void r_update_k(const float* __restrict__ Rw,
                           const float* __restrict__ dR,
                           const float* __restrict__ surprise,
                           float* __restrict__ Rn, int n) {
  int i = blockIdx.x * blockDim.x + threadIdx.x;
  if (i < n) {
    float scale = 0.01f * surprise[0] * (1.0f / (float)kB);
    float v = 0.999f * Rw[i] + scale * dR[i];
    Rn[i] = fminf(fmaxf(v, -3.0f), 3.0f);
  }
}

__global__ __launch_bounds__(256) void state_update_k(
    const float* __restrict__ core, const float* __restrict__ temporal,
    const float* __restrict__ hp, const float* __restrict__ g,
    const float* __restrict__ bb, float* __restrict__ h_t) {
  __shared__ float red[256];
  const int b = blockIdx.x, tid = threadIdx.x;
  float x[4], s = 0.0f, q = 0.0f;
#pragma unroll
  for (int i = 0; i < 4; ++i) {
    int e = tid + i * 256;
    size_t o = (size_t)b * kE + e;
    x[i] = tanhf(core[o] + 0.1f * temporal[o]);
    s += x[i]; q += x[i] * x[i];
  }
  s = block_sum(s, red);
  q = block_sum(q, red);
  float m = s * (1.0f / kE);
  float var = q * (1.0f / kE) - m * m;
  float rs = rsqrtf(var + 1e-5f);
#pragma unroll
  for (int i = 0; i < 4; ++i) {
    int e = tid + i * 256;
    size_t o = (size_t)b * kE + e;
    float ns = (x[i] - m) * rs * g[e] + bb[e];
    h_t[o] = 0.95f * hp[o] + 0.05f * ns;
  }
}

__global__ __launch_bounds__(256) void fused_ln_k(
    const float* __restrict__ emb, const float* __restrict__ h_t,
    const float* __restrict__ g, const float* __restrict__ bb,
    float* __restrict__ out) {
  __shared__ float red[256];
  const int b = blockIdx.x, tid = threadIdx.x;
  float x[16], s = 0.0f, q = 0.0f;
#pragma unroll
  for (int i = 0; i < 16; ++i) {
    int j = tid + i * 256;
    int seg = j >> 10, e = j & (kE - 1);
    size_t o = (size_t)b * kE + e;
    float ev = emb[o], hv = h_t[o];
    float v = (seg == 0) ? ev : (seg == 1) ? hv : (seg == 2) ? ev * hv : ev - hv;
    x[i] = v; s += v; q += v * v;
  }
  s = block_sum(s, red);
  q = block_sum(q, red);
  float m = s * (1.0f / kHD);
  float var = q * (1.0f / kHD) - m * m;
  float rs = rsqrtf(var + 1e-5f);
#pragma unroll
  for (int i = 0; i < 16; ++i) {
    int j = tid + i * 256;
    out[(size_t)b * kHD + j] = (x[i] - m) * rs * g[j] + bb[j];
  }
}

__global__ __launch_bounds__(256) void gelu_bias_ln_k(
    const float* __restrict__ pre, const float* __restrict__ bias,
    const float* __restrict__ g, const float* __restrict__ bb,
    float* __restrict__ out) {
  __shared__ float red[256];
  const int b = blockIdx.x, tid = threadIdx.x;
  float x[16], s = 0.0f, q = 0.0f;
#pragma unroll
  for (int i = 0; i < 16; ++i) {
    int j = tid + i * 256;
    float v = pre[(size_t)b * kHD + j] + bias[j];
    v = 0.5f * v * (1.0f + erff(v * 0.70710678118654752f));
    x[i] = v; s += v; q += v * v;
  }
  s = block_sum(s, red);
  q = block_sum(q, red);
  float m = s * (1.0f / kHD);
  float var = q * (1.0f / kHD) - m * m;
  float rs = rsqrtf(var + 1e-5f);
#pragma unroll
  for (int i = 0; i < 16; ++i) {
    int j = tid + i * 256;
    out[(size_t)b * kHD + j] = (x[i] - m) * rs * g[j] + bb[j];
  }
}

__global__ __launch_bounds__(256) void out_ln_k(
    const float* __restrict__ f2, const float* __restrict__ bias,
    const float* __restrict__ emb, const float* __restrict__ g,
    const float* __restrict__ bb, float* __restrict__ out) {
  __shared__ float red[256];
  const int b = blockIdx.x, tid = threadIdx.x;
  float x[4], s = 0.0f, q = 0.0f;
#pragma unroll
  for (int i = 0; i < 4; ++i) {
    int e = tid + i * 256;
    size_t o = (size_t)b * kE + e;
    x[i] = f2[o] + bias[e] + emb[o];
    s += x[i]; q += x[i] * x[i];
  }
  s = block_sum(s, red);
  q = block_sum(q, red);
  float m = s * (1.0f / kE);
  float var = q * (1.0f / kE) - m * m;
  float rs = rsqrtf(var + 1e-5f);
#pragma unroll
  for (int i = 0; i < 4; ++i) {
    int e = tid + i * 256;
    out[(size_t)b * kE + e] = (x[i] - m) * rs * g[e] + bb[e];
  }
}

// ---------------- orchestration ----------------
static inline dim3 gemm_grid(int M, int N) {
  return dim3((N + TBN - 1) / TBN, (M + TBM - 1) / TBM);
}

extern "C" void kernel_launch(void* const* d_in, const int* in_sizes, int n_in,
                              void* d_out, int out_size, void* d_ws, size_t ws_size,
                              hipStream_t stream) {
  const int*   token_ids = (const int*)d_in[0];
  const float* h_prev    = (const float*)d_in[1];
  const float* R_weight  = (const float*)d_in[2];
  const float* surprise  = (const float*)d_in[3];
  const float* emb_w     = (const float*)d_in[4];
  const float* W1        = (const float*)d_in[5];
  const float* b1        = (const float*)d_in[6];
  const float* W2        = (const float*)d_in[7];
  const float* b2        = (const float*)d_in[8];
  const float* rn_g      = (const float*)d_in[9];
  const float* rn_b      = (const float*)d_in[10];
  const float* fn_g      = (const float*)d_in[11];
  const float* fn_b      = (const float*)d_in[12];
  const float* p1_w      = (const float*)d_in[13];
  const float* p1_b      = (const float*)d_in[14];
  const float* pn_g      = (const float*)d_in[15];
  const float* pn_b      = (const float*)d_in[16];
  const float* p2_w      = (const float*)d_in[17];
  const float* p2_b      = (const float*)d_in[18];
  const float* on_g      = (const float*)d_in[19];
  const float* on_b      = (const float*)d_in[20];
  float* logits = (float*)d_out;

  // workspace carve-out
  float* ws       = (float*)d_ws;
  float* emb      = ws;                          // B*E
  float* g1       = emb      + (size_t)kB * kE;  // B*HH
  float* core     = g1       + (size_t)kB * kHH; // B*E
  float* xhat     = core     + (size_t)kB * kE;  // B*E
  float* epsb     = xhat     + (size_t)kB * kE;  // B*E
  float* dR       = epsb     + (size_t)kB * kE;  // E*E
  float* Rnew     = dR       + (size_t)kE * kE;  // E*E
  float* temporal = Rnew     + (size_t)kE * kE;  // B*E
  float* h_t      = temporal + (size_t)kB * kE;  // B*E
  float* fused    = h_t      + (size_t)kB * kE;  // B*4E
  float* h1p      = fused    + (size_t)kB * kHD; // B*4E
  float* h1       = h1p      + (size_t)kB * kHD; // B*4E
  float* f2       = h1       + (size_t)kB * kHD; // B*E
  float* outb     = f2       + (size_t)kB * kE;  // B*E
  unsigned short* emb_bf16 = (unsigned short*)(outb + (size_t)kB * kE); // V*E bf16
  (void)ws_size; (void)n_in; (void)in_sizes; (void)out_size;

  const dim3 blk(256);

  // 0. emb_w -> bf16 (L2-resident B panel for the LM head)
  {
    long long n8 = ((long long)kV * kE) / 8;
    convert_bf16_k<<<(unsigned)((n8 + 255) / 256), blk, 0, stream>>>(
        emb_w, emb_bf16, n8);
  }

  // 1. emb = l2norm(emb_w[token_ids])
  embed_l2norm_k<<<kB, blk, 0, stream>>>(token_ids, emb_w, emb);

  // 2. core = l2norm(tanh(emb@W1 + b1) @ W2 + b2)
  gemm_wmma_bf16<false, false, false, kHH><<<gemm_grid(kB, kHH), blk, 0, stream>>>(
      emb, W1, g1, kB, kHH, kE, kE, kHH);
  bias_tanh_k<<<(kB * kHH + 255) / 256, blk, 0, stream>>>(g1, b1, g1, kB * kHH);
  gemm_wmma_bf16<false, false, false, kE><<<gemm_grid(kB, kE), blk, 0, stream>>>(
      g1, W2, core, kB, kE, kHH, kHH, kE);
  bias_l2norm_k<<<kB, blk, 0, stream>>>(core, b2, core);

  // 3. delta-rule R update
  gemm_wmma_bf16<false, false, false, kE><<<gemm_grid(kB, kE), blk, 0, stream>>>(
      h_prev, R_weight, xhat, kB, kE, kE, kE, kE);
  sub_k<<<(kB * kE + 255) / 256, blk, 0, stream>>>(core, xhat, epsb, kB * kE);
  gemm_wmma_bf16<true, false, false, kE><<<gemm_grid(kE, kE), blk, 0, stream>>>(
      h_prev, epsb, dR, kE, kE, kB, kE, kE);            // hp^T @ eps
  r_update_k<<<(kE * kE + 255) / 256, blk, 0, stream>>>(R_weight, dR, surprise,
                                                        Rnew, kE * kE);

  // 4. temporal integration
  gemm_wmma_bf16<false, false, false, kE><<<gemm_grid(kB, kE), blk, 0, stream>>>(
      h_prev, Rnew, temporal, kB, kE, kE, kE, kE);
  state_update_k<<<kB, blk, 0, stream>>>(core, temporal, h_prev, rn_g, rn_b, h_t);

  // 5. fusion + MLP + residual
  fused_ln_k<<<kB, blk, 0, stream>>>(emb, h_t, fn_g, fn_b, fused);
  gemm_wmma_bf16<false, false, false, kHD><<<gemm_grid(kB, kHD), blk, 0, stream>>>(
      fused, p1_w, h1p, kB, kHD, kHD, kHD, kHD);
  gelu_bias_ln_k<<<kB, blk, 0, stream>>>(h1p, p1_b, pn_g, pn_b, h1);
  gemm_wmma_bf16<false, false, false, kE><<<gemm_grid(kB, kE), blk, 0, stream>>>(
      h1, p2_w, f2, kB, kE, kHD, kHD, kE);
  out_ln_k<<<kB, blk, 0, stream>>>(f2, p2_b, emb, on_g, on_b, outb);

  // 6. tied LM head: logits = out @ emb_w^T
  //    aligned main region (bf16 B, async global->LDS staging)
  gemm_wmma_lmhead<<<gemm_grid(kB, kVMain), blk, 0, stream>>>(
      outb, emb_bf16, logits);
  //    guarded edge: last 17 columns (fp32 path)
  gemm_wmma_bf16<false, true, true, kV><<<gemm_grid(kB, kV - kVMain), blk, 0, stream>>>(
      outb, emb_w + (size_t)kVMain * kE, logits + kVMain,
      kB, kV - kVMain, kE, kE, kE);
}
